// MMD_loss_80126909874743
// MI455X (gfx1250) — compile-verified, matrix operand order checked by reading, not run on hardware
//
#include <hip/hip_runtime.h>
#include <hip/hip_bf16.h>

typedef __attribute__((ext_vector_type(16))) __bf16 v16bf;
typedef __attribute__((ext_vector_type(8)))  __bf16 v8bf;
typedef __attribute__((ext_vector_type(8)))  float  v8f;

#define NROWS   512          // 256 source + 256 target rows
#define CDIM    256          // channel dim (K of the Gram GEMM)
#define HWTOT   256          // 16*16 spatial positions
#define CHUNK   32           // spatial positions staged per pass (ws = ~16.1 MB)

// ---------------------------------------------------------------------------
// Kernel 1: transpose [i][c][hw] fp32  ->  [hw_local][i][c] bf16 (hi + lo),
// and compute sq[hw_local][i] = sum_c x^2. Reads are fully coalesced
// (consecutive lanes -> consecutive hw).
// ---------------------------------------------------------------------------
__global__ __launch_bounds__(256)
void mmd_split_kernel(const float* __restrict__ src, const float* __restrict__ tgt,
                      __bf16* __restrict__ ahi, __bf16* __restrict__ alo,
                      float* __restrict__ sqout, int hw_base)
{
    int g        = blockIdx.x * 256 + threadIdx.x;   // 512*CHUNK threads total
    int hw_local = g & (CHUNK - 1);
    int i        = g >> 5;                            // CHUNK == 32
    if (i >= NROWS) return;
    int hw = hw_base + hw_local;

    const float* __restrict__ p =
        (i < 256) ? (src + (size_t)i * (CDIM * HWTOT))
                  : (tgt + (size_t)(i - 256) * (CDIM * HWTOT));

    __bf16* __restrict__ oh = ahi + ((size_t)hw_local * NROWS + i) * CDIM;
    __bf16* __restrict__ ol = alo + ((size_t)hw_local * NROWS + i) * CDIM;

    float sq = 0.0f;
#pragma unroll 4
    for (int c = 0; c < CDIM; ++c) {
        float x = p[(size_t)c * HWTOT + hw];
        sq += x * x;
        __bf16 h  = (__bf16)x;                 // round-to-nearest bf16
        float  lo = x - (float)h;              // residual for 3-term product
        oh[c] = h;
        ol[c] = (__bf16)lo;
    }
    sqout[hw_local * NROWS + i] = sq;
}

// Load one 16x32 bf16 A/B fragment: per lane two contiguous 16B runs
// (K = khi+0..7 and K = khi+16..23), matching the CDNA5 16-bit WMMA layout.
__device__ __forceinline__ v16bf load_frag(const __bf16* __restrict__ p)
{
    v16bf f;
    *((v8bf*)&f)       = *(const v8bf*)(p);
    *(((v8bf*)&f) + 1) = *(const v8bf*)(p + 16);
    return f;
}

__device__ __forceinline__ v8f wmma_bf16(v16bf a, v16bf b, v8f c)
{
    return __builtin_amdgcn_wmma_f32_16x16x32_bf16(
        /*neg_a=*/false, a, /*neg_b=*/false, b,
        /*c_mod=*/(short)0, c, /*reuse_a=*/false, /*reuse_b=*/false);
}

// ---------------------------------------------------------------------------
// Kernel 2: fused Gram (bf16x3 via v_wmma_f32_16x16x32_bf16) -> D -> sum of
// Gaussian kernels -> signed block reduction -> atomicAdd.
// grid.x = 36 upper-triangular 64x64 tiles of the 512x512 Gram (symmetry),
// grid.y = spatial position within chunk. 256 threads = 8 waves; each wave
// owns one 16-row A fragment and two 16-col accumulators.
// ---------------------------------------------------------------------------
__global__ __launch_bounds__(256)
void mmd_gram_kernel(const __bf16* __restrict__ ahi, const __bf16* __restrict__ alo,
                     const float* __restrict__ sq, float* __restrict__ out)
{
    int hw = blockIdx.y;

    // decode upper-triangular tile index -> (bi, bj), bi <= bj
    int t = blockIdx.x, bi = 0;
    while (t >= 8 - bi) { t -= 8 - bi; ++bi; }
    int bj = bi + t;
    int i0 = bi * 64, j0 = bj * 64;

    int tid  = threadIdx.x;
    int lane = tid & 31;
    int wave = tid >> 5;
    int tr   = wave & 3;          // tile row within 64x64 block (0..3)
    int tg   = wave >> 2;         // tile-column group (0..1), 2 cols each
    int lm   = lane & 15;
    int khi  = (lane & 16) ? 8 : 0;

    const __bf16* __restrict__ baseHi = ahi + (size_t)hw * NROWS * CDIM;
    const __bf16* __restrict__ baseLo = alo + (size_t)hw * NROWS * CDIM;

    const __bf16* aHiRow = baseHi + (size_t)(i0 + tr * 16 + lm) * CDIM + khi;
    const __bf16* aLoRow = baseLo + (size_t)(i0 + tr * 16 + lm) * CDIM + khi;

    const __bf16* bHiRow[2];
    const __bf16* bLoRow[2];
#pragma unroll
    for (int u = 0; u < 2; ++u) {
        int n = j0 + (tg * 2 + u) * 16 + lm;
        bHiRow[u] = baseHi + (size_t)n * CDIM + khi;
        bLoRow[u] = baseLo + (size_t)n * CDIM + khi;
    }

    v8f acc[2];
    acc[0] = (v8f)(0.0f);
    acc[1] = (v8f)(0.0f);

#pragma unroll 2
    for (int kk = 0; kk < CDIM; kk += 32) {
        v16bf ah = load_frag(aHiRow + kk);
        v16bf al = load_frag(aLoRow + kk);
#pragma unroll
        for (int u = 0; u < 2; ++u) {
            v16bf bh = load_frag(bHiRow[u] + kk);
            v16bf bl = load_frag(bLoRow[u] + kk);
            // bf16x3 split-product: hi*hi + hi*lo + lo*hi (f32 accumulate)
            acc[u] = wmma_bf16(ah, bh, acc[u]);
            acc[u] = wmma_bf16(ah, bl, acc[u]);
            acc[u] = wmma_bf16(al, bh, acc[u]);
        }
    }

    // ---- epilogue: D -> sum_b exp(-D/b), signed + symmetry-weighted -------
    const float invb[6] = {0.5f, 0.2f, 0.1f, 0.05f, 0.025f, 0.0125f};
    const float* __restrict__ sqhw = sq + hw * NROWS;

    float partial = 0.0f;
#pragma unroll
    for (int u = 0; u < 2; ++u) {
        int j  = j0 + (tg * 2 + u) * 16 + lm;
        float sqj = sqhw[j];
#pragma unroll
        for (int r = 0; r < 8; ++r) {
            int   i  = i0 + tr * 16 + ((lane & 16) ? (r + 8) : r);
            float d  = sqhw[i] + sqj - 2.0f * acc[u][r];
            float ks = 0.0f;
#pragma unroll
            for (int b = 0; b < 6; ++b) ks += __expf(-d * invb[b]);
            partial += ks;
        }
    }

    // block reduction
    __shared__ float red[256];
    red[tid] = partial;
    __syncthreads();
#pragma unroll
    for (int s = 128; s > 0; s >>= 1) {
        if (tid < s) red[tid] += red[tid + s];
        __syncthreads();
    }
    if (tid == 0) {
        float sgn    = ((i0 < 256) == (j0 < 256)) ? 1.0f : -1.0f;
        float weight = (bi == bj) ? 1.0f : 2.0f;           // symmetric Gram
        atomicAdd(out, red[0] * sgn * weight * (1.0f / 16777216.0f));
    }
}

extern "C" void kernel_launch(void* const* d_in, const int* in_sizes, int n_in,
                              void* d_out, int out_size, void* d_ws, size_t ws_size,
                              hipStream_t stream)
{
    const float* src = (const float*)d_in[0];   // (256,256,16,16) f32
    const float* tgt = (const float*)d_in[1];   // (256,256,16,16) f32
    float*       out = (float*)d_out;           // scalar f32

    // workspace layout (per 32-position chunk): hi plane, lo plane, sq
    const size_t planeElems = (size_t)CHUNK * NROWS * CDIM;   // 4 Mi bf16
    __bf16* ahi = (__bf16*)d_ws;
    __bf16* alo = ahi + planeElems;
    float*  sqb = (float*)(alo + planeElems);
    (void)in_sizes; (void)n_in; (void)out_size; (void)ws_size;

    hipMemsetAsync(out, 0, sizeof(float), stream);

    for (int ch = 0; ch < HWTOT / CHUNK; ++ch) {
        mmd_split_kernel<<<dim3((NROWS * CHUNK) / 256), 256, 0, stream>>>(
            src, tgt, ahi, alo, sqb, ch * CHUNK);
        mmd_gram_kernel<<<dim3(36, CHUNK), 256, 0, stream>>>(ahi, alo, sqb, out);
    }
}
